// Decoder_77326591197418
// MI455X (gfx1250) — matrix operand order repaired and checked by reading, not compiled, not run on hardware
//
#include <hip/hip_runtime.h>

static constexpr int kB    = 4;
static constexpr int kH    = 224;
static constexpr int kW    = 224;
static constexpr int kNT   = 257;
static constexpr int kNPAD = 272;   // 17*16
static constexpr int kCIN  = 1152;
static constexpr int kDF   = 512;
static constexpr int kDH   = 128;
static constexpr int kKPAD = 544;   // 3+512 padded to 17*32
static constexpr int kHW1  = kH * kW + 1;

typedef __attribute__((ext_vector_type(8)))  float        v8f;
typedef __attribute__((ext_vector_type(16))) __bf16       v16bf;
typedef __attribute__((ext_vector_type(4)))  unsigned int v4u;
typedef __attribute__((ext_vector_type(8)))  unsigned int v8u;

__device__ __forceinline__ unsigned short f2bf(float f) {
  union { __bf16 h; unsigned short u; } cv;
  cv.h = (__bf16)f;
  return cv.u;
}
__device__ __forceinline__ unsigned int pk2(float a, float b) {
  union { __bf16 h[2]; unsigned int u; } cv;
  cv.h[0] = (__bf16)a;
  cv.h[1] = (__bf16)b;
  return cv.u;
}
__device__ __forceinline__ v8f wmma_bf16(const unsigned int* a, const unsigned int* bb, v8f acc) {
  union { unsigned int u[8]; v16bf v; } A, Bv;
#pragma unroll
  for (int i = 0; i < 8; ++i) { A.u[i] = a[i]; Bv.u[i] = bb[i]; }
  return __builtin_amdgcn_wmma_f32_16x16x32_bf16(false, A.v, false, Bv.v,
                                                 (short)0, acc, false, false);
}
__device__ __forceinline__ int iclamp(int v, int lo, int hi) {
  return v < lo ? lo : (v > hi ? hi : v);
}
__device__ __forceinline__ unsigned lds_addr_of(const void* p) {
  return (unsigned)(uintptr_t)p;   // LDS aperture: low 32 bits are the LDS byte address
}

// Async global->LDS copy of one 16B chunk (ASYNCcnt path).
__device__ __forceinline__ void async_copy_b128(unsigned lds_byte_addr, const void* gaddr) {
  unsigned long long ga = (unsigned long long)(uintptr_t)gaddr;
  asm volatile("global_load_async_to_lds_b128 %0, %1, off"
               :: "v"(lds_byte_addr), "v"(ga)
               : "memory");
}
__device__ __forceinline__ void async_wait0() {
  asm volatile("s_wait_asynccnt 0x0" ::: "memory");
}

// Tensor Data Mover: 2-D tile global->LDS per CDNA5 ISA D# layout.
// data_size_code: 0=1B 1=2B 2=4B; stride in data_size units; LDS padding:
// pad_amount(+1) dwords inserted every 2^(pad_interval+1) stored dwords.
__device__ __forceinline__ void tdm_load_2d(unsigned lds_addr, const void* gaddr,
                                            unsigned data_size_code,
                                            unsigned tensor_d0, unsigned tensor_d1,
                                            unsigned tile_d0, unsigned tile_d1,
                                            unsigned long long d0_stride,
                                            unsigned pad_interval, unsigned pad_amount) {
  unsigned long long ga = (unsigned long long)(uintptr_t)gaddr;
  v4u g0;
  g0[0] = 0x1u;                                              // count=1, user D#
  g0[1] = lds_addr;                                          // lds_addr [63:32]
  g0[2] = (unsigned)(ga & 0xFFFFFFFFu);                      // global_addr lo
  g0[3] = (unsigned)((ga >> 32) & 0x01FFFFFFu) | (2u << 30); // ga hi + type=2
  v8u g1;
  g1[0] = (data_size_code << 16) | (1u << 20)                // pad_enable
        | (pad_interval << 22) | (pad_amount << 25);
  g1[1] = (tensor_d0 & 0xFFFFu) << 16;                       // tensor_dim0 lo16
  g1[2] = (tensor_d0 >> 16) | ((tensor_d1 & 0xFFFFu) << 16); // dim0 hi | dim1 lo
  g1[3] = (tensor_d1 >> 16) | ((tile_d0 & 0xFFFFu) << 16);   // dim1 hi | tile_dim0
  g1[4] = (tile_d1 & 0xFFFFu);                               // tile_dim1 (tile_dim2=0)
  g1[5] = (unsigned)(d0_stride & 0xFFFFFFFFu);               // dim0_stride lo32
  g1[6] = (unsigned)((d0_stride >> 32) & 0xFFFFu);           // dim0_stride hi16
  g1[7] = 0;                                                 // dim1_stride (unused, 2D)
  asm volatile("tensor_load_to_lds %0, %1" :: "s"(g0), "s"(g1) : "memory");
}

// ---------------- setup kernels ----------------

__global__ void cam_kernel(const float* __restrict__ c2w, const float* __restrict__ K,
                           float* __restrict__ cams) {
  int b = threadIdx.x;
  if (b >= kB) return;
  float m[12];
#pragma unroll
  for (int r = 0; r < 3; ++r)
#pragma unroll
    for (int c = 0; c < 4; ++c) {
      float v = c2w[b * 16 + r * 4 + c];
      if (c == 1 || c == 2) v = -v;          // flip[:3,1:3] = -1
      m[r * 4 + c] = v;
    }
  float a00 = m[0], a01 = m[1], a02 = m[2];
  float a10 = m[4], a11 = m[5], a12 = m[6];
  float a20 = m[8], a21 = m[9], a22 = m[10];
  float det = a00 * (a11 * a22 - a12 * a21) - a01 * (a10 * a22 - a12 * a20)
            + a02 * (a10 * a21 - a11 * a20);
  float id = 1.0f / det;
  float r00 = (a11 * a22 - a12 * a21) * id, r01 = (a02 * a21 - a01 * a22) * id,
        r02 = (a01 * a12 - a02 * a11) * id;
  float r10 = (a12 * a20 - a10 * a22) * id, r11 = (a00 * a22 - a02 * a20) * id,
        r12 = (a02 * a10 - a00 * a12) * id;
  float r20 = (a10 * a21 - a11 * a20) * id, r21 = (a01 * a20 - a00 * a21) * id,
        r22 = (a00 * a11 - a01 * a10) * id;
  float tx = m[3], ty = m[7], tz = m[11];
  float* o = cams + b * 16;
  o[0] = r00; o[1] = r01; o[2] = r02;
  o[3] = r10; o[4] = r11; o[5] = r12;
  o[6] = r20; o[7] = r21; o[8] = r22;
  o[9]  = -(r00 * tx + r01 * ty + r02 * tz);
  o[10] = -(r10 * tx + r11 * ty + r12 * tz);
  o[11] = -(r20 * tx + r21 * ty + r22 * tz);
  const float* Kb = K + b * 9;
  o[12] = Kb[0]; o[13] = Kb[4]; o[14] = Kb[2]; o[15] = Kb[5];
}

// One-shot bf16 conversion / packing of all reused operands.
__global__ void prep_w_kernel(const float* __restrict__ w1, const float* __restrict__ tokens,
                              const float* __restrict__ w2,
                              const float* __restrict__ s0w1, const float* __restrict__ s1w1,
                              const float* __restrict__ s0ow, const float* __restrict__ s1ow,
                              unsigned short* __restrict__ w1bf,
                              unsigned int*  __restrict__ tokp,
                              unsigned short* __restrict__ w2bf,
                              unsigned short* __restrict__ sw1bf0, unsigned short* __restrict__ sw1bf1,
                              unsigned short* __restrict__ owbf0, unsigned short* __restrict__ owbf1) {
  const int N_W1 = 2 * kDF * kCIN;          // 1024*1152 ushort
  const int N_TK = kB * (kCIN / 2) * kNPAD; // pair-packed uints
  const int N_W2 = kDF * 2 * kDF;
  const int N_SW = kDH * kKPAD;
  const int N_OW = 32 * kDH;
  const int total = N_W1 + N_TK + N_W2 + 2 * N_SW + 2 * N_OW;
  for (int i = blockIdx.x * blockDim.x + threadIdx.x; i < total;
       i += gridDim.x * blockDim.x) {
    int j = i;
    if (j < N_W1) { w1bf[j] = f2bf(w1[j]); continue; }
    j -= N_W1;
    if (j < N_TK) {
      int b = j / ((kCIN / 2) * kNPAD);
      int r = (j / kNPAD) % (kCIN / 2);     // k-pair index
      int n = j % kNPAD;
      float v0 = 0.f, v1 = 0.f;
      if (n < kNT) {
        const float* t = tokens + (size_t)b * kCIN * kNT;
        v0 = t[(size_t)(2 * r) * kNT + n];
        v1 = t[(size_t)(2 * r + 1) * kNT + n];
      }
      tokp[j] = pk2(v0, v1);
      continue;
    }
    j -= N_TK;
    if (j < N_W2) { w2bf[j] = f2bf(w2[j]); continue; }
    j -= N_W2;
    if (j < 2 * N_SW) {
      int s = j / N_SW, k = j % N_SW;
      int r = k / kKPAD, c = k % kKPAD;
      const float* src = s ? s1w1 : s0w1;
      unsigned short* dst = s ? sw1bf1 : sw1bf0;
      dst[k] = (c < 515) ? f2bf(src[r * 515 + c]) : (unsigned short)0;
      continue;
    }
    j -= 2 * N_SW;
    int s = j / N_OW, k = j % N_OW;
    int r = k / kDH, c = k % kDH;
    int rows = s ? 24 : 12;
    const float* src = s ? s1ow : s0ow;
    unsigned short* dst = s ? owbf1 : owbf0;
    dst[k] = (r < rows) ? f2bf(src[r * kDH + c]) : (unsigned short)0;
  }
}

__global__ void fill_u64_kernel(unsigned long long* __restrict__ p, size_t n) {
  size_t i = (size_t)blockIdx.x * blockDim.x + threadIdx.x;
  if (i < n) p[i] = ~0ull;
}

// ---------------- token MLP (WMMA GEMMs, async-LDS staging) ----------------
// GEMM1: feat1 = relu(w1 @ tokens + b1); M=1024 K=1152 N=272; out pair-packed bf16.
__global__ __launch_bounds__(256) void gemm1_kernel(
    const unsigned short* __restrict__ w1bf, const unsigned int* __restrict__ tokp,
    const float* __restrict__ b1, unsigned int* __restrict__ feat1p) {
  __shared__ unsigned int Ap[128][20];   // stride 20: 16B-aligned rows + conflict-free
  __shared__ unsigned int Bp[16][20];
  const int tid = threadIdx.x, lane = tid & 31, wave = tid >> 5;
  const int row = lane & 15, hi = lane >> 4;
  const int m0 = blockIdx.x * 128, n0 = blockIdx.y * 16, b = blockIdx.z;
  const unsigned ldsA = lds_addr_of(&Ap[0][0]);
  const unsigned ldsB = lds_addr_of(&Bp[0][0]);
  v8f acc = {};
  for (int kc = 0; kc < kCIN; kc += 32) {
#pragma unroll
    for (int it = 0; it < 2; ++it) {
      int idx = tid + it * 256;          // 512 chunks of 16B
      int r = idx >> 2, c4 = idx & 3;
      async_copy_b128(ldsA + (unsigned)((r * 20 + 4 * c4) * 4),
                      w1bf + (size_t)(m0 + r) * kCIN + kc + 8 * c4);
    }
    if (tid < 64) {                      // 64 chunks of 16B
      int jj = tid >> 2, c4 = tid & 3;
      async_copy_b128(ldsB + (unsigned)((jj * 20 + 4 * c4) * 4),
                      tokp + (size_t)(b * (kCIN / 2) + (kc >> 1) + jj) * kNPAD + n0 + 4 * c4);
    }
    async_wait0();
    __syncthreads();
    unsigned int af[8], bf[8];
#pragma unroll
    for (int r8 = 0; r8 < 8; ++r8) bf[r8] = Bp[hi * 8 + r8][row];
#pragma unroll
    for (int r8 = 0; r8 < 8; ++r8)
      af[r8] = Ap[wave * 16 + row][hi * 8 + r8];
    acc = wmma_bf16(af, bf, acc);
    __syncthreads();
  }
#pragma unroll
  for (int t = 0; t < 4; ++t) {
    int gm = m0 + wave * 16 + 8 * hi + 2 * t;   // even pair base
    float v0 = acc[2 * t]     + b1[gm];
    float v1 = acc[2 * t + 1] + b1[gm + 1];
    v0 = v0 > 0.f ? v0 : 0.f;
    v1 = v1 > 0.f ? v1 : 0.f;
    feat1p[((size_t)b * kDF + (gm >> 1)) * kNPAD + n0 + row] = pk2(v0, v1);
  }
}

// GEMM2: featT[b][n][m] = w2 @ feat1 + b2; M=512 K=1024 N=272; token-major f32 out.
__global__ __launch_bounds__(256) void gemm2_kernel(
    const unsigned short* __restrict__ w2bf, const unsigned int* __restrict__ feat1p,
    const float* __restrict__ b2, float* __restrict__ featT) {
  __shared__ unsigned int Ap[128][20];
  __shared__ unsigned int Bp[16][20];
  const int tid = threadIdx.x, lane = tid & 31, wave = tid >> 5;
  const int row = lane & 15, hi = lane >> 4;
  const int m0 = blockIdx.x * 128, n0 = blockIdx.y * 16, b = blockIdx.z;
  const unsigned ldsA = lds_addr_of(&Ap[0][0]);
  const unsigned ldsB = lds_addr_of(&Bp[0][0]);
  v8f acc = {};
  for (int kc = 0; kc < 2 * kDF; kc += 32) {
#pragma unroll
    for (int it = 0; it < 2; ++it) {
      int idx = tid + it * 256;
      int r = idx >> 2, c4 = idx & 3;
      async_copy_b128(ldsA + (unsigned)((r * 20 + 4 * c4) * 4),
                      w2bf + (size_t)(m0 + r) * 2 * kDF + kc + 8 * c4);
    }
    if (tid < 64) {
      int jj = tid >> 2, c4 = tid & 3;
      async_copy_b128(ldsB + (unsigned)((jj * 20 + 4 * c4) * 4),
                      feat1p + (size_t)(b * kDF + (kc >> 1) + jj) * kNPAD + n0 + 4 * c4);
    }
    async_wait0();
    __syncthreads();
    unsigned int af[8], bf[8];
#pragma unroll
    for (int r8 = 0; r8 < 8; ++r8) bf[r8] = Bp[hi * 8 + r8][row];
#pragma unroll
    for (int r8 = 0; r8 < 8; ++r8)
      af[r8] = Ap[wave * 16 + row][hi * 8 + r8];
    acc = wmma_bf16(af, bf, acc);
    __syncthreads();
  }
#pragma unroll
  for (int r8 = 0; r8 < 8; ++r8) {
    int gm = m0 + wave * 16 + r8 + 8 * hi;
    featT[((size_t)b * kNPAD + n0 + row) * kDF + gm] = acc[r8] + b2[gm];
  }
}

// ---------------- projection / z-buffer / gather ----------------

__global__ void proj_kernel(const float* __restrict__ pts, const float* __restrict__ cams,
                            unsigned long long* __restrict__ zbuf,
                            unsigned long long* __restrict__ keyArr, int* __restrict__ pixArr,
                            int N, int Ntot) {
  int p = blockIdx.x * blockDim.x + threadIdx.x;
  if (p >= Ntot) return;
  int b = p / N;
  const float* cm = cams + b * 16;
  float px = pts[p * 3 + 0], py = pts[p * 3 + 1], pz = pts[p * 3 + 2];
  float cx = cm[0] * px + cm[1] * py + cm[2] * pz + cm[9];
  float cy = cm[3] * px + cm[4] * py + cm[5] * pz + cm[10];
  float cz = cm[6] * px + cm[7] * py + cm[8] * pz + cm[11];
  float zs = (fabsf(cz) > 1e-8f) ? cz : 1e-8f;
  float ppx = cm[12] * cx / zs + cm[14];
  float ppy = cm[13] * cy / zs + cm[15];
  int ix = (int)floorf(ppx), iy = (int)floorf(ppy);
  bool valid = (cz > 1e-6f) && ix >= 0 && ix < kW && iy >= 0 && iy < kH;
  if (valid) {
    int pix = iy * kW + ix;
    unsigned long long key =
        ((unsigned long long)__float_as_uint(cz) << 32) | (unsigned int)p;
    atomicMin(&zbuf[(size_t)b * kHW1 + pix], key);
    keyArr[p] = key;
    pixArr[p] = pix;
  } else {
    pixArr[p] = -1;
  }
}

// One block (128 threads) per point: visibility + on-the-fly bilinear sample of the
// 16x16x512 grid (half-pixel, edge-clamped == jax.image.resize); emits pair-packed X.
__global__ __launch_bounds__(128) void gather_kernel(
    const float* __restrict__ pts, const float* __restrict__ featT,
    const unsigned long long* __restrict__ zbuf, const unsigned long long* __restrict__ keyArr,
    const int* __restrict__ pixArr, unsigned int* __restrict__ Xp, int N, int Npad) {
  int p = blockIdx.x, tid = threadIdx.x;
  int b = p / N;
  int pix = pixArr[p];
  bool visible = (pix >= 0) && (zbuf[(size_t)b * kHW1 + pix] == keyArr[p]);
  float w00 = 0, w01 = 0, w10 = 0, w11 = 0;
  const float *F00 = featT, *F01 = featT, *F10 = featT, *F11 = featT;
  if (visible) {
    int ix = pix % kW, iy = pix / kW;
    const float s = 16.0f / 224.0f;
    float u = (ix + 0.5f) * s - 0.5f;
    float v = (iy + 0.5f) * s - 0.5f;
    float fu = floorf(u), fv = floorf(v);
    float du = u - fu, dv = v - fv;
    int x0 = (int)fu, y0 = (int)fv;
    int gx0 = iclamp(x0, 0, 15), gx1 = iclamp(x0 + 1, 0, 15);
    int gy0 = iclamp(y0, 0, 15), gy1 = iclamp(y0 + 1, 0, 15);
    const float* base = featT + (size_t)b * kNPAD * kDF;
    F00 = base + (size_t)(1 + gy0 * 16 + gx0) * kDF;
    F01 = base + (size_t)(1 + gy0 * 16 + gx1) * kDF;
    F10 = base + (size_t)(1 + gy1 * 16 + gx0) * kDF;
    F11 = base + (size_t)(1 + gy1 * 16 + gx1) * kDF;
    w00 = (1 - dv) * (1 - du); w01 = (1 - dv) * du;
    w10 = dv * (1 - du);       w11 = dv * du;
  }
  auto valAt = [&](int k) -> float {
    if (k < 3) return pts[(size_t)p * 3 + k];
    if (k < 515) {
      if (!visible) return 0.f;
      int c = k - 3;
      return w00 * F00[c] + w01 * F01[c] + w10 * F10[c] + w11 * F11[c];
    }
    return 0.f;
  };
  for (int j = tid; j < kKPAD / 2; j += 128)
    Xp[(size_t)j * Npad + p] = pk2(valAt(2 * j), valAt(2 * j + 1));
}

// ---- fused stage MLP: TDM-staged WMMA(515->128) -> relu -> WMMA(128->f*3) -> tanh ----
__global__ __launch_bounds__(128) void stage_mlp_kernel(
    const unsigned short* __restrict__ Wbf,   // 128 x 544 bf16
    const unsigned short* __restrict__ owbf,  // 32  x 128 bf16 (rows zero-padded)
    const float* __restrict__ bs,             // 128
    const float* __restrict__ ob,             // f*3
    const unsigned int* __restrict__ Xp,      // (544/2) x Npad pair-packed bf16
    const float* __restrict__ pts_in,         // Npad x 3
    float* __restrict__ out,                  // Npad*f x 3
    int Npad, int f) {
  __shared__ unsigned int Wb[2][128 * 17];    // double-buffered W k-chunk, row stride 17
  __shared__ unsigned int Xl[272 * 17 + 4];   // resident X block, row stride 17 (TDM pad)
  __shared__ unsigned int Hp[64][17];
  const int tid = threadIdx.x, lane = tid & 31, wave = tid >> 5;
  const int row = lane & 15, hi = lane >> 4;
  const int col0 = blockIdx.x * 16;

  if (wave == 0) {   // TDM issue is wave-uniform (EXEC ignored by tensor ops)
    // Resident X block: 272 rows x 16 dwords, +1 pad dword per 16 -> stride 17.
    tdm_load_2d(lds_addr_of(Xl), Xp + col0,
                /*4B*/2, /*t_d0=*/16, /*t_d1=*/272, /*tile*/16, 272,
                (unsigned long long)(unsigned)Npad, /*pad*/3, 0);
    // First W chunk: 128 rows x 32 bf16 (16 dwords) -> stride 17.
    tdm_load_2d(lds_addr_of(&Wb[0][0]), Wbf,
                /*2B*/1, 32, 128, 32, 128, (unsigned long long)kKPAD, 3, 0);
    __builtin_amdgcn_s_wait_tensorcnt(0);
  }
  __syncthreads();

  v8f acc0 = {}, acc1 = {};
  for (int i = 0; i < kKPAD / 32; ++i) {      // 17 k-chunks
    if (wave == 0 && i + 1 < kKPAD / 32) {    // prefetch next W chunk into other buffer
      tdm_load_2d(lds_addr_of(&Wb[(i + 1) & 1][0]), Wbf + (i + 1) * 32,
                  1, 32, 128, 32, 128, (unsigned long long)kKPAD, 3, 0);
    }
    const unsigned int* Wc = &Wb[i & 1][0];
    unsigned int bfv[8], a0[8], a1[8];
#pragma unroll
    for (int r8 = 0; r8 < 8; ++r8)
      bfv[r8] = Xl[(i * 16 + hi * 8 + r8) * 17 + row];
#pragma unroll
    for (int r8 = 0; r8 < 8; ++r8) {
      int aidx = hi * 8 + r8;
      a0[r8] = Wc[((2 * wave)     * 16 + row) * 17 + aidx];
      a1[r8] = Wc[((2 * wave + 1) * 16 + row) * 17 + aidx];
    }
    acc0 = wmma_bf16(a0, bfv, acc0);
    acc1 = wmma_bf16(a1, bfv, acc1);
    if (wave == 0) __builtin_amdgcn_s_wait_tensorcnt(0);
    __syncthreads();
  }
  // relu + bias -> Hp (bf16 pairs along hidden dim)
#pragma unroll
  for (int s = 0; s < 2; ++s) {
    v8f acc = s ? acc1 : acc0;
    int sbase = 2 * wave + s;
    float h[8];
#pragma unroll
    for (int r8 = 0; r8 < 8; ++r8) {
      int rg = sbase * 16 + r8 + 8 * hi;
      float v = acc[r8] + bs[rg];
      h[r8] = v > 0.f ? v : 0.f;
    }
#pragma unroll
    for (int t = 0; t < 4; ++t)
      Hp[sbase * 8 + 4 * hi + t][row] = pk2(h[2 * t], h[2 * t + 1]);
  }
  __syncthreads();
  if (wave < 2) {
    v8f acc2 = {};
#pragma unroll
    for (int kc2 = 0; kc2 < 4; ++kc2) {
      unsigned int af[8], bf2[8];
#pragma unroll
      for (int r8 = 0; r8 < 8; ++r8) {
        int kb = kc2 * 32 + hi * 16 + 2 * r8;
        af[r8] = *(const unsigned int*)(owbf + (size_t)(wave * 16 + row) * kDH + kb);
      }
#pragma unroll
      for (int r8 = 0; r8 < 8; ++r8) bf2[r8] = Hp[kc2 * 16 + hi * 8 + r8][row];
      acc2 = wmma_bf16(af, bf2, acc2);
    }
    int colg = col0 + row;
#pragma unroll
    for (int r8 = 0; r8 < 8; ++r8) {
      int o = wave * 16 + r8 + 8 * hi;
      if (o < 3 * f) {
        float val = tanhf(acc2[r8] + ob[o]);        // RADIUS == 1.0
        int fi = o / 3, d = o - 3 * fi;
        out[((size_t)colg * f + fi) * 3 + d] = pts_in[(size_t)colg * 3 + d] + val;
      }
    }
  }
}

// ---------------- host launch ----------------

extern "C" void kernel_launch(void* const* d_in, const int* in_sizes, int n_in,
                              void* d_out, int out_size, void* d_ws, size_t ws_size,
                              hipStream_t stream) {
  const float* points = (const float*)d_in[0];
  const float* tokens = (const float*)d_in[1];
  const float* c2w    = (const float*)d_in[2];
  const float* Kin    = (const float*)d_in[3];
  const float* w1     = (const float*)d_in[4];
  const float* b1     = (const float*)d_in[5];
  const float* w2     = (const float*)d_in[6];
  const float* b2     = (const float*)d_in[7];
  const float* s0w1   = (const float*)d_in[8];
  const float* s0b1   = (const float*)d_in[9];
  const float* s0ow   = (const float*)d_in[10];
  const float* s0ob   = (const float*)d_in[11];
  const float* s1w1   = (const float*)d_in[12];
  const float* s1b1   = (const float*)d_in[13];
  const float* s1ow   = (const float*)d_in[14];
  const float* s1ob   = (const float*)d_in[15];
  float* out = (float*)d_out;

  char* ws = (char*)d_ws;
  size_t off = 0;
  auto walloc = [&](size_t bytes) -> char* {
    char* p = ws + off;
    off = (off + bytes + 255) & ~(size_t)255;
    return p;
  };
  float*              cams    = (float*)walloc(kB * 16 * sizeof(float));
  unsigned short*     w1bf    = (unsigned short*)walloc((size_t)2 * kDF * kCIN * 2);
  unsigned int*       tokp    = (unsigned int*)walloc((size_t)kB * (kCIN / 2) * kNPAD * 4);
  unsigned short*     w2bf    = (unsigned short*)walloc((size_t)kDF * 2 * kDF * 2);
  unsigned short*     sw1bf0  = (unsigned short*)walloc((size_t)kDH * kKPAD * 2);
  unsigned short*     sw1bf1  = (unsigned short*)walloc((size_t)kDH * kKPAD * 2);
  unsigned short*     owbf0   = (unsigned short*)walloc((size_t)32 * kDH * 2);
  unsigned short*     owbf1   = (unsigned short*)walloc((size_t)32 * kDH * 2);
  unsigned int*       feat1p  = (unsigned int*)walloc((size_t)kB * kDF * kNPAD * 4);
  float*              featT   = (float*)walloc((size_t)kB * kNPAD * kDF * sizeof(float));
  unsigned long long* zbuf    = (unsigned long long*)walloc((size_t)kB * kHW1 * 8);
  unsigned long long* keyArr  = (unsigned long long*)walloc((size_t)8192 * 8);
  int*                pixArr  = (int*)walloc((size_t)8192 * 4);
  unsigned int*       Xp      = (unsigned int*)walloc((size_t)(kKPAD / 2) * 8192 * 4);
  (void)ws_size; (void)in_sizes; (void)n_in; (void)out_size;

  cam_kernel<<<1, 32, 0, stream>>>(c2w, Kin, cams);
  {
    int total = 2 * kDF * kCIN + kB * (kCIN / 2) * kNPAD + kDF * 2 * kDF
              + 2 * kDH * kKPAD + 2 * 32 * kDH;
    prep_w_kernel<<<(total + 255) / 256, 256, 0, stream>>>(
        w1, tokens, w2, s0w1, s1w1, s0ow, s1ow,
        w1bf, tokp, w2bf, sw1bf0, sw1bf1, owbf0, owbf1);
  }
  gemm1_kernel<<<dim3(8, 17, kB), 256, 0, stream>>>(w1bf, tokp, b1, feat1p);
  gemm2_kernel<<<dim3(4, 17, kB), 256, 0, stream>>>(w2bf, feat1p, b2, featT);

  size_t nz = (size_t)kB * kHW1;

  // ---- stage 0: N=512/batch, f=4 ----
  {
    const int N = 512, Ntot = kB * N;     // 2048
    fill_u64_kernel<<<(unsigned)((nz + 255) / 256), 256, 0, stream>>>(zbuf, nz);
    proj_kernel<<<(Ntot + 255) / 256, 256, 0, stream>>>(points, cams, zbuf, keyArr, pixArr, N, Ntot);
    gather_kernel<<<Ntot, 128, 0, stream>>>(points, featT, zbuf, keyArr, pixArr, Xp, N, Ntot);
    stage_mlp_kernel<<<Ntot / 16, 128, 0, stream>>>(sw1bf0, owbf0, s0b1, s0ob, Xp, points, out, Ntot, 4);
  }
  // ---- stage 1: N=2048/batch, f=8 (input points = stage-0 output) ----
  {
    const int N = 2048, Ntot = kB * N;    // 8192
    const float* pts1 = out;              // (B, 2048, 3)
    float* out1 = out + (size_t)kB * 2048 * 3;
    fill_u64_kernel<<<(unsigned)((nz + 255) / 256), 256, 0, stream>>>(zbuf, nz);
    proj_kernel<<<(Ntot + 255) / 256, 256, 0, stream>>>(pts1, cams, zbuf, keyArr, pixArr, N, Ntot);
    gather_kernel<<<Ntot, 128, 0, stream>>>(pts1, featT, zbuf, keyArr, pixArr, Xp, N, Ntot);
    stage_mlp_kernel<<<Ntot / 16, 128, 0, stream>>>(sw1bf1, owbf1, s1b1, s1ob, Xp, pts1, out1, Ntot, 8);
  }
}
